// InferenceLTPMBlock_42030549959153
// MI455X (gfx1250) — matrix operand order repaired and checked by reading, not compiled, not run on hardware
//
#include <hip/hip_runtime.h>
#include <hip/hip_bf16.h>
#include <math.h>

// ---------------------------------------------------------------------------
// InferenceLTPMBlock for MI455X (gfx1250, wave32, WMMA 16x16x32 f16,
// async HBM->LDS staging pipeline on the GEMMs)
// ---------------------------------------------------------------------------
#define NTOK   2048
#define CH     768
#define NHEAD  12
#define HDIM   64
#define HMLP   3072
#define QKVN   (3*CH)      // 2304
#define QKCOLS (2*CH)      // 1536 (q|k packed), v stored transposed separately

typedef __attribute__((ext_vector_type(16))) _Float16 v16h;
typedef __attribute__((ext_vector_type(8)))  _Float16 v8h;
typedef __attribute__((ext_vector_type(8)))  float    v8f;

// ---- gfx1250 async copy: 16B global -> LDS per lane, tracked by ASYNCcnt ---
__device__ __forceinline__ void async_copy16(unsigned ldsAddr, const void* gptr) {
  asm volatile("global_load_async_to_lds_b128 %0, %1, off"
               :: "v"(ldsAddr), "v"(gptr) : "memory");
}
#if __has_builtin(__builtin_amdgcn_s_wait_asynccnt)
#define WAIT_ASYNC(n) __builtin_amdgcn_s_wait_asynccnt(n)
#else
#define WAIT_ASYNC(n) asm volatile("s_wait_asynccnt %0" :: "i"(n) : "memory")
#endif

// ---- WMMA fragment helpers (16-bit layouts per CDNA5 ISA 7.12.2) ----------
// A (16x32, MxK): lane L holds row M=L&15; K chunks (L>>4)*8..+7 and 16+(L>>4)*8..+7
__device__ __forceinline__ v16h load_frag_a(const _Float16* p, int ld, int lane) {
  int r = lane & 15, hh = lane >> 4;
  const _Float16* base = p + (size_t)r * ld + hh * 8;
  v8h lo = *(const v8h*)(base);
  v8h hi = *(const v8h*)(base + 16);
  v16h out;
#pragma unroll
  for (int i = 0; i < 8; i++) { out[i] = lo[i]; out[i + 8] = hi[i]; }
  return out;
}
// B (32x16, KxN): lane L holds column N=L&15; K chunks (L>>4)*8..+7 and 16+(L>>4)*8..+7
// p points at an array stored [n][k] row-major (i.e. B^T row-major, k contiguous).
__device__ __forceinline__ v16h load_frag_b(const _Float16* p, int ld, int lane) {
  int n = lane & 15, hh = lane >> 4;
  const _Float16* base = p + (size_t)n * ld + hh * 8;
  v8h lo = *(const v8h*)(base);
  v8h hi = *(const v8h*)(base + 16);
  v16h out;
#pragma unroll
  for (int i = 0; i < 8; i++) { out[i] = lo[i]; out[i + 8] = hi[i]; }
  return out;
}
__device__ __forceinline__ v8f wmma_f16(v16h a, v16h b, v8f c) {
  return __builtin_amdgcn_wmma_f32_16x16x32_f16(false, a, false, b, (short)0, c, false, false);
}

// ---------------------------------------------------------------------------
// Generic WMMA GEMM:  out[m][n] = act( sum_k A[m][k]*B[n][k] + bias[n] ) (+res)
// block = 256 (8 waves), block tile 64(M) x 128(N), wave tile 32x32.
// K-step 32 double-buffered through LDS via global_load_async_to_lds_b128.
// OMODE: 0 = f32 store, 1 = f32 + residual, 2 = f16 store, 3 = qkv split
// ---------------------------------------------------------------------------
template<int OMODE, bool BIAS, bool GELU>
__global__ void gemm_wmma(const _Float16* __restrict__ A, int lda,
                          const _Float16* __restrict__ B, int ldb,
                          const float* __restrict__ bias,
                          const float* __restrict__ res, int ldr,
                          float* __restrict__ outF,
                          _Float16* __restrict__ outH,
                          _Float16* __restrict__ vtOut,
                          int ldo, int K) {
  __shared__ __align__(16) _Float16 aBuf[2][64 * 32];   // 2 x 4KB
  __shared__ __align__(16) _Float16 bBuf[2][128 * 32];  // 2 x 8KB

  const int tid  = threadIdx.x;
  const int lane = tid & 31;
  const int w    = tid >> 5;
  const int wm   = w & 1, wn = w >> 1;
  const int rowB = blockIdx.y * 64;    // block M origin
  const int colB = blockIdx.x * 128;   // block N origin

  // per-thread staging coordinates (16B per async op)
  const int ra = tid >> 2, ca = (tid & 3) * 8;  // A: 64 rows x 32 cols, 1 op
  const unsigned aL0 = (unsigned)(size_t)&aBuf[0][0] + (unsigned)tid * 16u;
  const unsigned aL1 = (unsigned)(size_t)&aBuf[1][0] + (unsigned)tid * 16u;
  const unsigned bL0 = (unsigned)(size_t)&bBuf[0][0];
  const unsigned bL1 = (unsigned)(size_t)&bBuf[1][0];

  auto stage = [&](int k0, int buf) {
    async_copy16((buf ? aL1 : aL0), A + (size_t)(rowB + ra) * lda + k0 + ca);
    const unsigned bl = buf ? bL1 : bL0;
    const int e0 = tid, e1 = tid + 256;  // B: 128 rows x 32 cols, 2 ops
    async_copy16(bl + (unsigned)e0 * 16u,
                 B + (size_t)(colB + (e0 >> 2)) * ldb + k0 + (e0 & 3) * 8);
    async_copy16(bl + (unsigned)e1 * 16u,
                 B + (size_t)(colB + (e1 >> 2)) * ldb + k0 + (e1 & 3) * 8);
  };

  v8f acc[2][2] = {};
  const int T = K >> 5;
  stage(0, 0);
  for (int t = 0; t < T; ++t) {
    const int buf = t & 1;
    if (t + 1 < T) { stage((t + 1) << 5, buf ^ 1); WAIT_ASYNC(3); }
    else           { WAIT_ASYNC(0); }
    __syncthreads();

    const _Float16* aS = &aBuf[buf][0];
    const _Float16* bS = &bBuf[buf][0];
    v16h a0 = load_frag_a(aS + (wm * 32) * 32,        32, lane);
    v16h a1 = load_frag_a(aS + (wm * 32 + 16) * 32,   32, lane);
    v16h b0 = load_frag_b(bS + (wn * 32) * 32,        32, lane);
    v16h b1 = load_frag_b(bS + (wn * 32 + 16) * 32,   32, lane);
    acc[0][0] = wmma_f16(a0, b0, acc[0][0]);
    acc[0][1] = wmma_f16(a0, b1, acc[0][1]);
    acc[1][0] = wmma_f16(a1, b0, acc[1][0]);
    acc[1][1] = wmma_f16(a1, b1, acc[1][1]);

    __syncthreads();  // everyone done reading buf before it is refilled
  }

  const int row0 = rowB + wm * 32;
  const int col0 = colB + wn * 32;
  const int hl = lane >> 4, cn = lane & 15;
#pragma unroll
  for (int i = 0; i < 2; i++) {
#pragma unroll
    for (int j = 0; j < 2; j++) {
#pragma unroll
      for (int r = 0; r < 8; r++) {
        int m = row0 + i * 16 + r + hl * 8;
        int n = col0 + j * 16 + cn;
        float v = acc[i][j][r];
        if (BIAS) v += bias[n];
        if (GELU) v = 0.5f * v * (1.0f + erff(v * 0.70710678118654752f));
        if (OMODE == 0) {
          outF[(size_t)m * ldo + n] = v;
        } else if (OMODE == 1) {
          outF[(size_t)m * ldo + n] = v + res[(size_t)m * ldr + n];
        } else if (OMODE == 2) {
          outH[(size_t)m * ldo + n] = (_Float16)v;
        } else {  // qkv split: cols [0,1536) -> q|k row-major; [1536,2304) -> v transposed
          if (n < QKCOLS) outH[(size_t)m * QKCOLS + n] = (_Float16)v;
          else            vtOut[(size_t)(n - QKCOLS) * NTOK + m] = (_Float16)v;
        }
      }
    }
  }
}

// ---------------------------------------------------------------------------
// Attention pass 1: per (head, 16-query tile) row max M and sum L via WMMA QK^T
// block = 128 (4 waves, one q-tile each); grid = 12 heads * 32
// ---------------------------------------------------------------------------
__global__ void attn_pass1(const _Float16* __restrict__ qk,
                           const float* __restrict__ logsz,
                           float* __restrict__ Mv, float* __restrict__ Lv) {
  const int lane = threadIdx.x & 31;
  const int w    = threadIdx.x >> 5;
  const int h    = blockIdx.x >> 5;
  const int q0   = ((blockIdx.x & 31) * 4 + w) * 16;
  const int hl = lane >> 4, cn = lane & 15;

  const _Float16* Qb = qk + (size_t)q0 * QKCOLS + h * HDIM;
  v16h aq0 = load_frag_a(Qb,      QKCOLS, lane);
  v16h aq1 = load_frag_a(Qb + 32, QKCOLS, lane);

  float mr[8], lr[8];
#pragma unroll
  for (int r = 0; r < 8; r++) { mr[r] = -3.0e38f; lr[r] = 0.0f; }

  for (int kk = 0; kk < NTOK; kk += 32) {
    const _Float16* Kb = qk + (size_t)kk * QKCOLS + CH + h * HDIM;
    v16h b00 = load_frag_b(Kb,                    QKCOLS, lane);
    v16h b01 = load_frag_b(Kb + 32,               QKCOLS, lane);
    v16h b10 = load_frag_b(Kb + 16 * QKCOLS,      QKCOLS, lane);
    v16h b11 = load_frag_b(Kb + 16 * QKCOLS + 32, QKCOLS, lane);
    v8f S0 = {}, S1 = {};
    S0 = wmma_f16(aq0, b00, S0); S0 = wmma_f16(aq1, b01, S0);
    S1 = wmma_f16(aq0, b10, S1); S1 = wmma_f16(aq1, b11, S1);
    float ls0 = logsz[kk + cn], ls1 = logsz[kk + 16 + cn];
#pragma unroll
    for (int r = 0; r < 8; r++) {
      float v0 = S0[r] * 0.125f + ls0;
      float nm = fmaxf(mr[r], v0);
      lr[r] = lr[r] * __expf(mr[r] - nm) + __expf(v0 - nm);
      mr[r] = nm;
      float v1 = S1[r] * 0.125f + ls1;
      nm = fmaxf(mr[r], v1);
      lr[r] = lr[r] * __expf(mr[r] - nm) + __expf(v1 - nm);
      mr[r] = nm;
    }
  }
  // reduce across the 16 lanes of each half (same q rows)
#pragma unroll
  for (int r = 0; r < 8; r++) {
    for (int off = 1; off < 16; off <<= 1) {
      float mo = __shfl_xor(mr[r], off, 32);
      float lo = __shfl_xor(lr[r], off, 32);
      float nm = fmaxf(mr[r], mo);
      lr[r] = lr[r] * __expf(mr[r] - nm) + lo * __expf(mo - nm);
      mr[r] = nm;
    }
  }
  if (cn == 0) {
#pragma unroll
    for (int r = 0; r < 8; r++) {
      int q = q0 + r + hl * 8;
      Mv[h * NTOK + q] = mr[r];
      Lv[h * NTOK + q] = lr[r];
    }
  }
}

// ---------------------------------------------------------------------------
// Attention pass 2: normalized probs -> O = P@V (WMMA), importance col-sums
// ---------------------------------------------------------------------------
__global__ void attn_pass2(const _Float16* __restrict__ qk,
                           const _Float16* __restrict__ vt,
                           const float* __restrict__ logsz,
                           const float* __restrict__ Mv, const float* __restrict__ Lv,
                           _Float16* __restrict__ o_h, float* __restrict__ imp) {
  __shared__ float impLds[NTOK];
  __shared__ __align__(32) _Float16 pLds[4][16 * 32];
  for (int i = threadIdx.x; i < NTOK; i += 128) impLds[i] = 0.0f;
  __syncthreads();

  const int lane = threadIdx.x & 31;
  const int w    = threadIdx.x >> 5;
  const int h    = blockIdx.x >> 5;
  const int q0   = ((blockIdx.x & 31) * 4 + w) * 16;
  const int hl = lane >> 4, cn = lane & 15;

  const _Float16* Qb = qk + (size_t)q0 * QKCOLS + h * HDIM;
  v16h aq0 = load_frag_a(Qb,      QKCOLS, lane);
  v16h aq1 = load_frag_a(Qb + 32, QKCOLS, lane);

  float mrow[8], rln[8];
#pragma unroll
  for (int r = 0; r < 8; r++) {
    int q = q0 + r + hl * 8;
    mrow[r] = Mv[h * NTOK + q];
    rln[r]  = 1.0f / Lv[h * NTOK + q];
  }
  v8f accO[4] = {};

  for (int kk = 0; kk < NTOK; kk += 32) {
    const _Float16* Kb = qk + (size_t)kk * QKCOLS + CH + h * HDIM;
    v16h b00 = load_frag_b(Kb,                    QKCOLS, lane);
    v16h b01 = load_frag_b(Kb + 32,               QKCOLS, lane);
    v16h b10 = load_frag_b(Kb + 16 * QKCOLS,      QKCOLS, lane);
    v16h b11 = load_frag_b(Kb + 16 * QKCOLS + 32, QKCOLS, lane);
    v8f S0 = {}, S1 = {};
    S0 = wmma_f16(aq0, b00, S0); S0 = wmma_f16(aq1, b01, S0);
    S1 = wmma_f16(aq0, b10, S1); S1 = wmma_f16(aq1, b11, S1);

    float ls0 = logsz[kk + cn], ls1 = logsz[kk + 16 + cn];
    float cs0 = 0.0f, cs1 = 0.0f;
#pragma unroll
    for (int r = 0; r < 8; r++) {
      float p0 = __expf(S0[r] * 0.125f + ls0 - mrow[r]) * rln[r];
      float p1 = __expf(S1[r] * 0.125f + ls1 - mrow[r]) * rln[r];
      cs0 += p0; cs1 += p1;
      int ql = r + hl * 8;
      pLds[w][ql * 32 + cn]      = (_Float16)p0;
      pLds[w][ql * 32 + 16 + cn] = (_Float16)p1;
    }
    atomicAdd(&impLds[kk + cn], cs0);
    atomicAdd(&impLds[kk + 16 + cn], cs1);

    v16h pf = load_frag_a(&pLds[w][0], 32, lane);  // same-wave LDS, in-order
#pragma unroll
    for (int db = 0; db < 4; db++) {
      const _Float16* Vb = vt + (size_t)(h * HDIM + db * 16) * NTOK + kk;
      v16h bv = load_frag_b(Vb, NTOK, lane);
      accO[db] = wmma_f16(pf, bv, accO[db]);
    }
  }
#pragma unroll
  for (int db = 0; db < 4; db++) {
#pragma unroll
    for (int r = 0; r < 8; r++) {
      int q = q0 + r + hl * 8;
      o_h[(size_t)q * CH + h * HDIM + db * 16 + cn] = (_Float16)accO[db][r];
    }
  }
  __syncthreads();
  for (int i = threadIdx.x; i < NTOK; i += 128) atomicAdd(&imp[i], impLds[i]);
}

// ---------------------------------------------------------------------------
// Small helper kernels
// ---------------------------------------------------------------------------
__global__ void f32_to_f16_k(const float* __restrict__ s, _Float16* __restrict__ d, int n) {
  int i = blockIdx.x * 256 + threadIdx.x;
  if (i < n) d[i] = (_Float16)s[i];
}
__global__ void zero_f32_k(float* p, int n) {
  int i = blockIdx.x * 256 + threadIdx.x;
  if (i < n) p[i] = 0.0f;
}

// LayerNorm (block=256, one row each). If cnt!=nullptr: rows >= cnt[2] -> zeros.
__global__ void layernorm_k(const float* __restrict__ x,
                            const float* __restrict__ wgt, const float* __restrict__ bst,
                            _Float16* __restrict__ out,
                            const int* __restrict__ cnt,
                            const float* __restrict__ szv, float* __restrict__ logsz) {
  int row = blockIdx.x, tid = threadIdx.x;
  if (cnt && row >= cnt[2]) {
    for (int j = tid; j < CH; j += 256) out[(size_t)row * CH + j] = (_Float16)0.0f;
    return;
  }
  __shared__ float s1[256], s2[256];
  float vals[3]; float a = 0.0f, b = 0.0f;
#pragma unroll
  for (int e = 0; e < 3; e++) {
    float v = x[(size_t)row * CH + tid + e * 256];
    vals[e] = v; a += v; b += v * v;
  }
  s1[tid] = a; s2[tid] = b; __syncthreads();
  for (int off = 128; off; off >>= 1) {
    if (tid < off) { s1[tid] += s1[tid + off]; s2[tid] += s2[tid + off]; }
    __syncthreads();
  }
  float mean = s1[0] * (1.0f / CH);
  float var  = s2[0] * (1.0f / CH) - mean * mean;
  float rstd = rsqrtf(var + 1e-5f);
#pragma unroll
  for (int e = 0; e < 3; e++) {
    int c = tid + e * 256;
    out[(size_t)row * CH + c] = (_Float16)((vals[e] - mean) * rstd * wgt[c] + bst[c]);
  }
  if (logsz && tid == 0) logsz[row] = logf(szv[row]);
}

// metric[n][d] = mean over heads of K
__global__ void metric_k(const _Float16* __restrict__ qk, float* __restrict__ metric) {
  int n = blockIdx.x, d = threadIdx.x;  // 64 threads
  float s = 0.0f;
  for (int h = 0; h < NHEAD; h++) s += (float)qk[(size_t)n * QKCOLS + CH + h * HDIM + d];
  metric[n * HDIM + d] = s * (1.0f / NHEAD);
}

// Prune scan: single block, 1024 threads, inclusive scan over 2048 keep flags.
__global__ void prune_scan_k(const float* __restrict__ imp, const float* __restrict__ thr,
                             int* __restrict__ gidx, int* __restrict__ counts) {
  __shared__ int sA[NTOK], sB[NTOK];
  int tid = threadIdx.x;
  float t = thr[0] * (float)(NHEAD * NTOK);  // imp holds sum over (h,q)
  int f0 = imp[tid] > t ? 1 : 0;
  int f1 = imp[tid + 1024] > t ? 1 : 0;
  sA[tid] = f0; sA[tid + 1024] = f1;
  __syncthreads();
  int* src = sA; int* dst = sB;
  for (int off = 1; off < NTOK; off <<= 1) {
    for (int e = 0; e < 2; e++) {
      int i = tid + e * 1024;
      int v = src[i];
      if (i >= off) v += src[i - off];
      dst[i] = v;
    }
    __syncthreads();
    int* tmp = src; src = dst; dst = tmp;
  }
  if (f0) gidx[src[tid] - 1] = tid;
  if (f1) gidx[src[tid + 1024] - 1] = tid + 1024;
  if (tid == 0) counts[0] = src[NTOK - 1];
}

// Gather kept rows: x*(size), normalized metric, size
__global__ void gather_k(const float* __restrict__ xres, const float* __restrict__ metric,
                         const float* __restrict__ szv, const int* __restrict__ gidx,
                         const int* __restrict__ counts,
                         float* __restrict__ xk, float* __restrict__ metrick,
                         float* __restrict__ sizek) {
  int i = blockIdx.x, tid = threadIdx.x;
  int Np = counts[0];
  if (i >= Np) {
    for (int j = tid; j < CH; j += 256) xk[(size_t)i * CH + j] = 0.0f;
    if (tid < HDIM) metrick[i * HDIM + tid] = 0.0f;
    if (tid == 0) sizek[i] = 1.0f;
    return;
  }
  int s = gidx[i];
  float sz = szv[s];
  for (int j = tid; j < CH; j += 256) xk[(size_t)i * CH + j] = xres[(size_t)s * CH + j] * sz;
  __shared__ float red[64];
  float v = 0.0f;
  if (tid < HDIM) { v = metric[s * HDIM + tid]; red[tid] = v * v; }
  __syncthreads();
  for (int off = 32; off >= 1; off >>= 1) {
    if (tid < off && tid + off < HDIM) red[tid] += red[tid + off];
    __syncthreads();
  }
  float nrm = rsqrtf(red[0]);
  if (tid < HDIM) metrick[i * HDIM + tid] = v * nrm;
  if (tid == 0) sizek[i] = sz;
}

// Per-src-row max/argmax over dst similarity (cosine). Row 0 forced to -inf.
__global__ void merge_scores_k(const float* __restrict__ metrick, const int* __restrict__ counts,
                               float* __restrict__ node_max, int* __restrict__ node_idx) {
  int s = blockIdx.x, tid = threadIdx.x;
  int Np = counts[0], Ns = (Np + 1) >> 1, Nd = Np >> 1;
  if (s >= Ns) {
    if (tid == 0) { node_max[s] = -__builtin_huge_valf(); node_idx[s] = 0; }
    return;
  }
  __shared__ float av[64];
  __shared__ float bmax[256];
  __shared__ int   bidx[256];
  if (tid < 64) av[tid] = metrick[(size_t)(2 * s) * HDIM + tid];
  __syncthreads();
  float best = -__builtin_huge_valf(); int bi = 0;
  if (s != 0) {
    for (int t = tid; t < Nd; t += 256) {
      const float* b = metrick + (size_t)(2 * t + 1) * HDIM;
      float dot = 0.0f;
#pragma unroll
      for (int d = 0; d < 64; d++) dot += av[d] * b[d];
      if (dot > best) { best = dot; bi = t; }
    }
  }
  bmax[tid] = best; bidx[tid] = bi; __syncthreads();
  for (int off = 128; off; off >>= 1) {
    if (tid < off) {
      if (bmax[tid + off] > bmax[tid] ||
          (bmax[tid + off] == bmax[tid] && bidx[tid + off] < bidx[tid])) {
        bmax[tid] = bmax[tid + off]; bidx[tid] = bidx[tid + off];
      }
    }
    __syncthreads();
  }
  if (tid == 0) {
    node_max[s] = (s == 0) ? -__builtin_huge_valf() : bmax[0];
    node_idx[s] = bidx[0];
  }
}

// Scan of unmerged src rows -> ordered list; counts[1]=Nu, counts[2]=Nu+Nd
__global__ void unm_scan_k(const float* __restrict__ node_max, const float* __restrict__ mthr,
                           int* __restrict__ counts, int* __restrict__ unm_list) {
  __shared__ int sA[1024], sB[1024];
  int tid = threadIdx.x;
  int Np = counts[0], Ns = (Np + 1) >> 1, Nd = Np >> 1;
  float mt = mthr[0];
  int flag = (tid < Ns && !(node_max[tid] > mt)) ? 1 : 0;
  sA[tid] = flag; __syncthreads();
  int* src = sA; int* dst = sB;
  for (int off = 1; off < 1024; off <<= 1) {
    int v = src[tid];
    if (tid >= off) v += src[tid - off];
    dst[tid] = v;
    __syncthreads();
    int* tmp = src; src = dst; dst = tmp;
  }
  if (flag) unm_list[src[tid] - 1] = tid;
  if (tid == 0) { counts[1] = src[1023]; counts[2] = src[1023] + Nd; }
}

// Build merged layout: rows [0,Nu) = unmerged src, [Nu,Nu+Nd) = dst, rest zero
__global__ void merge_init_k(const float* __restrict__ xk, const float* __restrict__ sizek,
                             const int* __restrict__ counts, const int* __restrict__ unm_list,
                             float* __restrict__ x2, float* __restrict__ size2) {
  int i = blockIdx.x, tid = threadIdx.x;
  int Nu = counts[1], N2 = counts[2];
  int srow;
  if (i < Nu) srow = 2 * unm_list[i];
  else if (i < N2) srow = 2 * (i - Nu) + 1;
  else {
    for (int j = tid; j < CH; j += 256) x2[(size_t)i * CH + j] = 0.0f;
    if (tid == 0) size2[i] = 1.0f;
    return;
  }
  for (int j = tid; j < CH; j += 256) x2[(size_t)i * CH + j] = xk[(size_t)srow * CH + j];
  if (tid == 0) size2[i] = sizek[srow];
}

// Scatter-add merged src rows into their dst rows
__global__ void merge_scatter_k(const float* __restrict__ xk, const float* __restrict__ sizek,
                                const int* __restrict__ counts,
                                const float* __restrict__ node_max, const int* __restrict__ node_idx,
                                const float* __restrict__ mthr,
                                float* __restrict__ x2, float* __restrict__ size2) {
  int s = blockIdx.x, tid = threadIdx.x;
  int Np = counts[0], Ns = (Np + 1) >> 1, Nu = counts[1];
  if (s >= Ns) return;
  if (!(node_max[s] > mthr[0])) return;
  int drow = Nu + node_idx[s];
  for (int j = tid; j < CH; j += 256)
    atomicAdd(&x2[(size_t)drow * CH + j], xk[(size_t)(2 * s) * CH + j]);
  if (tid == 0) atomicAdd(&size2[drow], sizek[2 * s]);
}

// xfinal = x2 / size2 (padded rows -> 0)
__global__ void finalize_div_k(const float* __restrict__ x2, const float* __restrict__ size2,
                               const int* __restrict__ counts, float* __restrict__ xfinal) {
  int i = blockIdx.x, tid = threadIdx.x;
  int N2 = counts[2];
  float inv = (i < N2) ? (1.0f / size2[i]) : 0.0f;
  for (int j = tid; j < CH; j += 256)
    xfinal[(size_t)i * CH + j] = x2[(size_t)i * CH + j] * inv;
}

// Pack outputs: d_out = [x (N2*768) | size (N2)]
__global__ void pack_out_k(const float* __restrict__ xout, const float* __restrict__ size2,
                           const int* __restrict__ counts, float* __restrict__ out) {
  int i = blockIdx.x * 256 + threadIdx.x;
  int N2 = counts[2];
  int nx = N2 * CH;
  if (i < nx)            out[i] = xout[i];
  else if (i < nx + N2)  out[i] = size2[i - nx];
}

// ---------------------------------------------------------------------------
extern "C" void kernel_launch(void* const* d_in, const int* in_sizes, int n_in,
                              void* d_out, int out_size, void* d_ws, size_t ws_size,
                              hipStream_t stream) {
  const float* x_in    = (const float*)d_in[0];
  const float* size_in = (const float*)d_in[1];
  const float* qkv_w   = (const float*)d_in[2];
  const float* proj_w  = (const float*)d_in[3];
  const float* proj_b  = (const float*)d_in[4];
  const float* n1w     = (const float*)d_in[5];
  const float* n1b     = (const float*)d_in[6];
  const float* n2w     = (const float*)d_in[7];
  const float* n2b     = (const float*)d_in[8];
  const float* fc1_w   = (const float*)d_in[9];
  const float* fc1_b   = (const float*)d_in[10];
  const float* fc2_w   = (const float*)d_in[11];
  const float* fc2_b   = (const float*)d_in[12];
  const float* pthr    = (const float*)d_in[13];
  const float* mthr    = (const float*)d_in[14];
  float* out = (float*)d_out;

  // ---- workspace carve-up ----
  char* wsp = (char*)d_ws;
  auto alloc = [&](size_t bytes) { void* p = (void*)wsp; wsp += (bytes + 255) & ~size_t(255); return p; };
  _Float16* qkvw_h  = (_Float16*)alloc((size_t)QKVN * CH * 2);
  _Float16* projw_h = (_Float16*)alloc((size_t)CH * CH * 2);
  _Float16* fc1w_h  = (_Float16*)alloc((size_t)HMLP * CH * 2);
  _Float16* fc2w_h  = (_Float16*)alloc((size_t)CH * HMLP * 2);
  _Float16* xn_h    = (_Float16*)alloc((size_t)NTOK * CH * 2);
  float*    logsz   = (float*)   alloc((size_t)NTOK * 4);
  _Float16* qk_h    = (_Float16*)alloc((size_t)NTOK * QKCOLS * 2);
  _Float16* vt_h    = (_Float16*)alloc((size_t)CH * NTOK * 2);
  float*    Mv      = (float*)   alloc((size_t)NHEAD * NTOK * 4);
  float*    Lv      = (float*)   alloc((size_t)NHEAD * NTOK * 4);
  float*    imp     = (float*)   alloc((size_t)NTOK * 4);
  _Float16* o_h     = (_Float16*)alloc((size_t)NTOK * CH * 2);
  float*    xres    = (float*)   alloc((size_t)NTOK * CH * 4);
  float*    metric  = (float*)   alloc((size_t)NTOK * HDIM * 4);
  int*      gidx    = (int*)     alloc((size_t)NTOK * 4);
  int*      counts  = (int*)     alloc(256);
  float*    xk      = (float*)   alloc((size_t)NTOK * CH * 4);
  float*    metrick = (float*)   alloc((size_t)NTOK * HDIM * 4);
  float*    sizek   = (float*)   alloc((size_t)NTOK * 4);
  float*    node_max= (float*)   alloc(1024 * 4);
  int*      node_idx= (int*)     alloc(1024 * 4);
  int*      unm_list= (int*)     alloc(1024 * 4);
  float*    x2      = (float*)   alloc((size_t)NTOK * CH * 4);
  float*    size2   = (float*)   alloc((size_t)NTOK * 4);
  float*    xfinal  = (float*)   alloc((size_t)NTOK * CH * 4);
  _Float16* ln2_h   = (_Float16*)alloc((size_t)NTOK * CH * 2);
  _Float16* h1_h    = (_Float16*)alloc((size_t)NTOK * HMLP * 2);
  float*    xout    = (float*)   alloc((size_t)NTOK * CH * 4);
  (void)ws_size; (void)in_sizes; (void)n_in; (void)out_size;

  // 1) weights -> f16
  auto cvt = [&](const float* s, _Float16* d, int n) {
    f32_to_f16_k<<<(n + 255) / 256, 256, 0, stream>>>(s, d, n);
  };
  cvt(qkv_w, qkvw_h, QKVN * CH);
  cvt(proj_w, projw_h, CH * CH);
  cvt(fc1_w, fc1w_h, HMLP * CH);
  cvt(fc2_w, fc2w_h, CH * HMLP);

  // 2) LN1 (+ log(size))
  layernorm_k<<<NTOK, 256, 0, stream>>>(x_in, n1w, n1b, xn_h, nullptr, size_in, logsz);

  // 3) QKV GEMM (split epilogue: q|k row-major, v transposed)
  gemm_wmma<3, false, false><<<dim3(QKVN / 128, NTOK / 64), 256, 0, stream>>>(
      xn_h, CH, qkvw_h, CH, nullptr, nullptr, 0, nullptr, qk_h, vt_h, 0, CH);

  // 4) attention (two-pass flash) + importance
  zero_f32_k<<<(NTOK + 255) / 256, 256, 0, stream>>>(imp, NTOK);
  attn_pass1<<<NHEAD * 32, 128, 0, stream>>>(qk_h, logsz, Mv, Lv);
  attn_pass2<<<NHEAD * 32, 128, 0, stream>>>(qk_h, vt_h, logsz, Mv, Lv, o_h, imp);

  // 5) metric = mean_h(K)
  metric_k<<<NTOK, 64, 0, stream>>>(qk_h, metric);

  // 6) out-proj + residual -> xres
  gemm_wmma<1, true, false><<<dim3(CH / 128, NTOK / 64), 256, 0, stream>>>(
      o_h, CH, projw_h, CH, proj_b, x_in, CH, xres, nullptr, nullptr, CH, CH);

  // 7) prune + compact
  prune_scan_k<<<1, 1024, 0, stream>>>(imp, pthr, gidx, counts);
  gather_k<<<NTOK, 256, 0, stream>>>(xres, metric, size_in, gidx, counts, xk, metrick, sizek);

  // 8) bipartite merge
  merge_scores_k<<<1024, 256, 0, stream>>>(metrick, counts, node_max, node_idx);
  unm_scan_k<<<1, 1024, 0, stream>>>(node_max, mthr, counts, unm_list);
  merge_init_k<<<NTOK, 256, 0, stream>>>(xk, sizek, counts, unm_list, x2, size2);
  merge_scatter_k<<<1024, 256, 0, stream>>>(xk, sizek, counts, node_max, node_idx, mthr, x2, size2);
  finalize_div_k<<<NTOK, 256, 0, stream>>>(x2, size2, counts, xfinal);

  // 9) MLP: LN2 -> fc1(+GELU) -> fc2(+residual)
  layernorm_k<<<NTOK, 256, 0, stream>>>(xfinal, n2w, n2b, ln2_h, counts, nullptr, nullptr);
  gemm_wmma<2, true, true><<<dim3(HMLP / 128, NTOK / 64), 256, 0, stream>>>(
      ln2_h, CH, fc1w_h, CH, fc1_b, nullptr, 0, nullptr, h1_h, nullptr, HMLP, CH);
  gemm_wmma<1, true, false><<<dim3(CH / 128, NTOK / 64), 256, 0, stream>>>(
      h1_h, HMLP, fc2w_h, HMLP, fc2_b, xfinal, CH, xout, nullptr, nullptr, CH, HMLP);

  // 10) pack outputs [x | size]
  pack_out_k<<<(NTOK * (CH + 1) + 255) / 256, 256, 0, stream>>>(xout, size2, counts, out);
}
